// GCN_LSTM_66503273611821
// MI455X (gfx1250) — compile-verified, matrix operand order checked by reading, not run on hardware
//
#include <hip/hip_runtime.h>
#include <hip/hip_bf16.h>

// ---------------------------------------------------------------------------
// GCN(Cheb K=2) -> 3x LSTM -> MLP head, MI455X (gfx1250, wave32, WMMA bf16)
// All GEMMs use V_WMMA_F32_16X16X32_BF16 (f32 accumulate).
// Weights pre-packed "fragment-major" so each lane loads its 16 fragment
// elements as one contiguous 32B run (2x b128) instead of 16 scalar loads.
// fc1 stages B tiles into LDS with global_load_async_to_lds_b128 (ASYNCcnt).
// ---------------------------------------------------------------------------

#define NCH     62
#define FDIM    16
#define TDIM    32
#define BDIM    256
#define NNODES  (BDIM * NCH)      // 15872
#define EDIM    200000
#define SEQ     512               // TDIM * FDIM
#define FC1K    131072            // SEQ * 256

typedef __bf16 v16bf __attribute__((ext_vector_type(16)));
typedef __bf16 v8bf  __attribute__((ext_vector_type(8)));
typedef float  v8f   __attribute__((ext_vector_type(8)));
typedef float  v4f   __attribute__((ext_vector_type(4)));

#if defined(__gfx1250__) && __has_builtin(__builtin_amdgcn_global_load_async_to_lds_b128) && __has_builtin(__builtin_amdgcn_s_wait_asynccnt)
#define USE_ASYNC_COPY 1
// Probe-derived parameter type: pointer to 16B int vector, AS1 (global) / AS3 (LDS).
typedef int v4i_vs __attribute__((vector_size(16)));
typedef __attribute__((address_space(1))) v4i_vs as1_v4i;
typedef __attribute__((address_space(3))) v4i_vs as3_v4i;
#else
#define USE_ASYNC_COPY 0
#endif

// K index inside a 16x32 bf16 A/B fragment for element e (0..15) of this lane.
// (ISA 7.12.2: VGPR0..3 hold K = 8*half+0..7, VGPR4..7 hold K = 16+8*half+0..7)
__device__ __forceinline__ int kmap(int e, int half) {
  return (e < 8) ? (8 * half + e) : (16 + 8 * half + (e - 8));
}

__device__ __forceinline__ v8bf ld8(const __bf16* p) { return *(const v8bf*)p; }

__device__ __forceinline__ v16bf cat16(v8bf lo, v8bf hi) {
  return __builtin_shufflevector(lo, hi, 0, 1, 2, 3, 4, 5, 6, 7, 8, 9, 10, 11, 12, 13,
                                 14, 15);
}

__device__ __forceinline__ float sigm(float x) { return 1.0f / (1.0f + __expf(-x)); }

// ---------------------------------------------------------------------------
// Graph preprocessing kernels
// ---------------------------------------------------------------------------

// x [N,F,T] f32 -> xt [T,N,F] f32
__global__ void k_transpose_x(const float* __restrict__ x, float* __restrict__ xt) {
  int idx = blockIdx.x * blockDim.x + threadIdx.x;
  const int total = NNODES * FDIM * TDIM;
  if (idx >= total) return;
  int t = idx % TDIM;
  int f = (idx / TDIM) % FDIM;
  int n = idx / (TDIM * FDIM);
  xt[((size_t)t * NNODES + n) * FDIM + f] = x[idx];
}

// deg[t][src] += w  (segment_sum by src)
__global__ void k_deg(const long long* __restrict__ ei, const float* __restrict__ w,
                      float* __restrict__ deg) {
  int idx = blockIdx.x * blockDim.x + threadIdx.x;
  const int total = TDIM * EDIM;
  if (idx >= total) return;
  int t = idx / EDIM, e = idx % EDIM;
  long long src = ei[(size_t)t * 2 * EDIM + e];
  atomicAdd(&deg[(size_t)t * NNODES + src], w[(size_t)t * EDIM + e]);
}

// deg -> d^{-1/2} in place
__global__ void k_dinv(float* __restrict__ deg) {
  int idx = blockIdx.x * blockDim.x + threadIdx.x;
  if (idx >= TDIM * NNODES) return;
  float d = deg[idx];
  deg[idx] = (d > 0.0f) ? rsqrtf(d) : 0.0f;
}

// tx[t][dst][:] += norm * feat[t][src][:],  norm = -dinv[src]*w*dinv[dst]
__global__ void k_scatter(const long long* __restrict__ ei, const float* __restrict__ w,
                          const float* __restrict__ dinv, const float* __restrict__ feat,
                          float* __restrict__ tx) {
  int idx = blockIdx.x * blockDim.x + threadIdx.x;
  const int total = TDIM * EDIM;
  if (idx >= total) return;
  int t = idx / EDIM, e = idx % EDIM;
  long long src = ei[(size_t)t * 2 * EDIM + e];
  long long dst = ei[(size_t)t * 2 * EDIM + EDIM + e];
  float norm = -dinv[(size_t)t * NNODES + src] * w[(size_t)t * EDIM + e] *
                dinv[(size_t)t * NNODES + dst];
  const float* fs = &feat[((size_t)t * NNODES + src) * FDIM];
  float* fd = &tx[((size_t)t * NNODES + dst) * FDIM];
#pragma unroll
  for (int f = 0; f < FDIM; ++f) atomicAdd(&fd[f], norm * fs[f]);
}

// ---------------------------------------------------------------------------
// Fragment-major weight packing (f32 -> bf16).
// Layout: tile (nt, c) -> out[((nt*ctiles + c)*32 + lane)*16 + e], so each lane
// reads its whole 16-element fragment as one contiguous 32B run.
// ---------------------------------------------------------------------------

// Generic B [K][N] row-major, logical width Npad (zero-padded cols).
__global__ void k_pack_frag(const float* __restrict__ w, __bf16* __restrict__ out,
                            int K, int N, int Npad) {
  int idx = blockIdx.x * blockDim.x + threadIdx.x;
  int ctiles = K >> 5;
  int total = (Npad >> 4) * ctiles * 512;
  if (idx >= total) return;
  int e = idx & 15;
  int lane = (idx >> 4) & 31;
  int tile = idx >> 9;
  int c = tile % ctiles;
  int nt = tile / ctiles;
  int half = lane >> 4, r = lane & 15;
  int k = c * 32 + kmap(e, half);
  int n = nt * 16 + r;
  out[idx] = (__bf16)((n < N) ? w[(size_t)k * N + n] : 0.0f);
}

// LSTM virtual B[Kpad][4H] = [wih^T ; 0-pad rows ; whh^T], fragment-major.
__global__ void k_pack_lstm_frag(const float* __restrict__ wih,
                                 const float* __restrict__ whh, __bf16* __restrict__ out,
                                 int in, int inPad, int H) {
  int idx = blockIdx.x * blockDim.x + threadIdx.x;
  int fourH = 4 * H;
  int ctiles = (inPad + H) >> 5;
  int total = (fourH >> 4) * ctiles * 512;
  if (idx >= total) return;
  int e = idx & 15;
  int lane = (idx >> 4) & 31;
  int tile = idx >> 9;
  int c = tile % ctiles;
  int nt = tile / ctiles;
  int half = lane >> 4, r = lane & 15;
  int k = c * 32 + kmap(e, half);
  int n = nt * 16 + r;
  float v = 0.0f;
  if (k < in)                           v = wih[(size_t)n * in + k];
  else if (k >= inPad && k < inPad + H) v = whh[(size_t)n * H + (k - inPad)];
  out[idx] = (__bf16)v;
}

// Cheb: single 32x16 tile, B = [w0 ; w1], fragment-major.
__global__ void k_pack_cheb(const float* __restrict__ w0, const float* __restrict__ w1,
                            __bf16* __restrict__ out) {
  int idx = blockIdx.x * blockDim.x + threadIdx.x;
  if (idx >= 512) return;
  int e = idx & 15;
  int lane = (idx >> 4) & 31;
  int half = lane >> 4, r = lane & 15;
  int k = kmap(e, half);
  out[idx] = (__bf16)((k < 16) ? w0[k * 16 + r] : w1[(k - 16) * 16 + r]);
}

// ---------------------------------------------------------------------------
// ChebConv GEMM: one wave per (t, 16-row node tile); single 16x16x32 WMMA
// computes x@w0 + tx@w1 (K-concat). mode 1: leaky_relu -> h1 (f32 [T,N,F]);
// mode 2: -> xcb bf16 [B,SEQ,64] (stride-64 padded, cols 62..63 zero).
// ---------------------------------------------------------------------------
__global__ void k_cheb(const float* __restrict__ X, const float* __restrict__ TX,
                       const __bf16* __restrict__ W, const float* __restrict__ bias,
                       int mode, float* __restrict__ h1, __bf16* __restrict__ xcb) {
  int wid  = blockIdx.x * (blockDim.x >> 5) + (threadIdx.x >> 5);
  int lane = threadIdx.x & 31;
  int half = lane >> 4, r = lane & 15;
  const int mtiles = NNODES / 16;  // 992
  int t = wid / mtiles, mt = wid % mtiles;
  int m0 = mt * 16;
  int row = m0 + r;

  const float* xrow  = &X[((size_t)t * NNODES + row) * FDIM];
  const float* txrow = &TX[((size_t)t * NNODES + row) * FDIM];
  v4f x0 = *(const v4f*)(xrow + 8 * half);
  v4f x1 = *(const v4f*)(xrow + 8 * half + 4);
  v4f t0 = *(const v4f*)(txrow + 8 * half);
  v4f t1 = *(const v4f*)(txrow + 8 * half + 4);
  v16bf a;
#pragma unroll
  for (int e = 0; e < 4; ++e) {
    a[e]      = (__bf16)x0[e];
    a[e + 4]  = (__bf16)x1[e];
    a[e + 8]  = (__bf16)t0[e];
    a[e + 12] = (__bf16)t1[e];
  }
  v16bf b = *(const v16bf*)&W[lane * 16];

  v8f acc;
  float bv = bias[r];
#pragma unroll
  for (int j = 0; j < 8; ++j) acc[j] = bv;
  acc = __builtin_amdgcn_wmma_f32_16x16x32_bf16(false, a, false, b, (short)0, acc,
                                                false, false);
#pragma unroll
  for (int j = 0; j < 8; ++j) {
    int orow = m0 + j + 8 * half;   // C-layout: VGPR j -> M = j (+8 for hi lanes)
    float v = acc[j];
    if (mode == 1) {
      v = (v > 0.0f) ? v : 0.01f * v;
      h1[((size_t)t * NNODES + orow) * FDIM + r] = v;
    } else {
      int bb = orow / NCH, ch = orow % NCH;
      int s = t * FDIM + r;         // xc[b, t*16+f, ch] = h2[t, b*62+ch, f]
      xcb[((size_t)bb * SEQ + s) * 64 + ch] = (__bf16)v;
    }
  }
}

// ---------------------------------------------------------------------------
// Fused LSTM layer: persistent grid, one wave per 16(batch) x 16(hidden) tile.
// g = [x_t | h_{t-1}] @ [wih|whh]^T + b, all 4 gates via WMMA; cell state c
// lives in the wave's v8f accumulator layout; h double-buffered in global;
// grid-wide atomic barrier (s_sleep backoff) between steps.  Kpad = inPad + H.
// ---------------------------------------------------------------------------
__global__ void k_lstm(const __bf16* __restrict__ xc, const __bf16* __restrict__ Wb,
                       const float* __restrict__ bias, __bf16* __restrict__ hbuf,
                       __bf16* __restrict__ hseq, unsigned* __restrict__ sync,
                       int inPad, int H, int nblocks) {
  __shared__ unsigned s_gen;
  if (threadIdx.x == 0) s_gen = 0;
  __syncthreads();

  int lane = threadIdx.x & 31;
  int half = lane >> 4, r = lane & 15;
  int wid = blockIdx.x * (blockDim.x >> 5) + (threadIdx.x >> 5);
  int htiles = H >> 4;
  int ctiles = (inPad + H) >> 5;
  int mt = wid / htiles, ht = wid % htiles;
  int m0 = mt * 16;
  int n = ht * 16 + r;
  int row = m0 + r;

  float biasv[4];
#pragma unroll
  for (int g = 0; g < 4; ++g) biasv[g] = bias[g * H + n];

  v8f c;
#pragma unroll
  for (int j = 0; j < 8; ++j) c[j] = 0.0f;

  for (int t = 0; t < SEQ; ++t) {
    const __bf16* hb_r = hbuf + (size_t)(t & 1) * BDIM * H;
    __bf16* hb_w       = hbuf + (size_t)((t + 1) & 1) * BDIM * H;
    const __bf16* xrow = xc + ((size_t)row * SEQ + t) * inPad;
    const __bf16* hrow = hb_r + (size_t)row * H;

    v8f acc[4];
#pragma unroll
    for (int g = 0; g < 4; ++g)
#pragma unroll
      for (int j = 0; j < 8; ++j) acc[g][j] = biasv[g];

    for (int cc = 0; cc < ctiles; ++cc) {
      int k0 = cc << 5;
      int ks0 = k0 + 8 * half;        // both 8-runs never straddle regions:
      int ks1 = k0 + 16 + 8 * half;   // inPad, H are multiples of 8
      v8bf lo = (ks0 < inPad) ? ld8(xrow + ks0) : ld8(hrow + (ks0 - inPad));
      v8bf hi = (ks1 < inPad) ? ld8(xrow + ks1) : ld8(hrow + (ks1 - inPad));
      v16bf a = cat16(lo, hi);
#pragma unroll
      for (int g = 0; g < 4; ++g) {
        v16bf b = *(const v16bf*)(Wb +
            ((size_t)((g * htiles + ht) * ctiles + cc) * 32 + lane) * 16);
        acc[g] = __builtin_amdgcn_wmma_f32_16x16x32_bf16(false, a, false, b, (short)0,
                                                         acc[g], false, false);
      }
    }

#pragma unroll
    for (int j = 0; j < 8; ++j) {
      float gi = acc[0][j], gf = acc[1][j], gg = acc[2][j], go = acc[3][j];
      float cn = sigm(gf) * c[j] + sigm(gi) * tanhf(gg);
      c[j] = cn;
      float h = sigm(go) * tanhf(cn);
      int orow = m0 + j + 8 * half;
      __bf16 hb = (__bf16)h;
      hb_w[(size_t)orow * H + n] = hb;
      hseq[((size_t)orow * SEQ + t) * H + n] = hb;
    }

    // ---- grid-wide barrier ----
    __threadfence();
    __syncthreads();
    if (threadIdx.x == 0) {
      unsigned target = s_gen + 1;
      unsigned a0 = atomicAdd(&sync[0], 1u);
      if (a0 == (unsigned)nblocks - 1u) {
        atomicExch(&sync[0], 0u);
        __threadfence();
        atomicAdd(&sync[1], 1u);
      }
      while (atomicAdd(&sync[1], 0u) < target) __builtin_amdgcn_s_sleep(1);
      s_gen = target;
    }
    __syncthreads();
    __threadfence();  // acquire: refresh cached h before next read
  }
}

// ---------------------------------------------------------------------------
// fc1: M=256, N=256, K=131072. Split-K (grid.y), fragment-major B tiles (1KB)
// staged into LDS via global_load_async_to_lds_b128 (fallback: uint4 copy),
// 8 waves/block each computing 2 M-tiles for one N-tile. Partials f32.
// ---------------------------------------------------------------------------
__global__ void k_fc1(const __bf16* __restrict__ A, const __bf16* __restrict__ Bw,
                      float* __restrict__ part, int KS) {
  __shared__ __align__(16) __bf16 lb[512];
  int nt = blockIdx.x;           // 0..15
  int ks = blockIdx.y;           // 0..KS-1
  int lane = threadIdx.x & 31;
  int half = lane >> 4, r = lane & 15;
  int w = threadIdx.x >> 5;      // wave 0..7
  int mt0 = w * 2;
  int n = nt * 16 + r;
  const int ctiles = FC1K / 32;  // 4096
  int cchunk = ctiles / KS;
  int cbeg = ks * cchunk, cend = cbeg + cchunk;

  const __bf16* arow0 = A + (size_t)(mt0 * 16 + r) * FC1K;
  const __bf16* arow1 = A + (size_t)((mt0 + 1) * 16 + r) * FC1K;

  v8f acc0, acc1;
#pragma unroll
  for (int j = 0; j < 8; ++j) { acc0[j] = 0.0f; acc1[j] = 0.0f; }

  for (int cc = cbeg; cc < cend; ++cc) {
    const __bf16* tileB = Bw + ((size_t)(nt * ctiles + cc) * 512);
    __syncthreads();
#if USE_ASYNC_COPY
    if (threadIdx.x < 64) {
      __builtin_amdgcn_global_load_async_to_lds_b128(
          (as1_v4i*)(tileB + threadIdx.x * 8),
          (as3_v4i*)&lb[threadIdx.x * 8], 0, 0);
    }
    __builtin_amdgcn_s_wait_asynccnt(0);
#else
    if (threadIdx.x < 64)
      ((uint4*)lb)[threadIdx.x] = ((const uint4*)tileB)[threadIdx.x];
#endif
    if (cc + 1 < cend)
      __builtin_prefetch(Bw + ((size_t)(nt * ctiles + cc + 1) * 512), 0, 1);
    __syncthreads();

    int k0 = cc * 32;
    v16bf b = *(const v16bf*)&lb[lane * 16];
    v16bf a0 = cat16(ld8(arow0 + k0 + 8 * half), ld8(arow0 + k0 + 16 + 8 * half));
    v16bf a1 = cat16(ld8(arow1 + k0 + 8 * half), ld8(arow1 + k0 + 16 + 8 * half));
    acc0 = __builtin_amdgcn_wmma_f32_16x16x32_bf16(false, a0, false, b, (short)0, acc0,
                                                   false, false);
    acc1 = __builtin_amdgcn_wmma_f32_16x16x32_bf16(false, a1, false, b, (short)0, acc1,
                                                   false, false);
  }
#pragma unroll
  for (int j = 0; j < 8; ++j) {
    int row0 = mt0 * 16 + j + 8 * half;
    part[((size_t)ks * 256 + row0) * 256 + n] = acc0[j];
    part[((size_t)ks * 256 + row0 + 16) * 256 + n] = acc1[j];
  }
}

__global__ void k_fc1_reduce(const float* __restrict__ part, const float* __restrict__ bias,
                             __bf16* __restrict__ z, int KS) {
  int idx = blockIdx.x * blockDim.x + threadIdx.x;
  if (idx >= 256 * 256) return;
  int m = idx >> 8, n = idx & 255;
  float s = bias[n];
  for (int ks = 0; ks < KS; ++ks) s += part[((size_t)ks * 256 + m) * 256 + n];
  z[idx] = (__bf16)fmaxf(s, 0.0f);
}

// ---------------------------------------------------------------------------
// Generic small GEMM: one wave per 16x16 tile. A[M][lda] bf16 row-major,
// B fragment-major. act: 1 = ReLU. Writes bf16 and/or f32, cols < Nreal only.
// ---------------------------------------------------------------------------
__global__ void k_gemm(const __bf16* __restrict__ A, int lda,
                       const __bf16* __restrict__ Bw,
                       const float* __restrict__ bias, int act,
                       __bf16* __restrict__ outb, float* __restrict__ outf, int ldo,
                       int ntiles, int Nreal, int K) {
  int wid = blockIdx.x * (blockDim.x >> 5) + (threadIdx.x >> 5);
  int lane = threadIdx.x & 31;
  int half = lane >> 4, r = lane & 15;
  int mt = wid / ntiles, nt = wid % ntiles;
  int m0 = mt * 16;
  int n = nt * 16 + r;
  int ctiles = K >> 5;
  const __bf16* arow = A + (size_t)(m0 + r) * lda;

  v8f acc;
  float bv = (n < Nreal) ? bias[n] : 0.0f;
#pragma unroll
  for (int j = 0; j < 8; ++j) acc[j] = bv;

  for (int cc = 0; cc < ctiles; ++cc) {
    int k0 = cc << 5;
    v16bf a = cat16(ld8(arow + k0 + 8 * half), ld8(arow + k0 + 16 + 8 * half));
    v16bf b = *(const v16bf*)(Bw + ((size_t)(nt * ctiles + cc) * 32 + lane) * 16);
    acc = __builtin_amdgcn_wmma_f32_16x16x32_bf16(false, a, false, b, (short)0, acc,
                                                  false, false);
  }
#pragma unroll
  for (int j = 0; j < 8; ++j) {
    int row = m0 + j + 8 * half;
    float v = acc[j];
    if (act == 1) v = fmaxf(v, 0.0f);
    if (n < Nreal) {
      if (outb) outb[(size_t)row * ldo + n] = (__bf16)v;
      if (outf) outf[(size_t)row * ldo + n] = v;
    }
  }
}

// ---------------------------------------------------------------------------
// Host orchestration
// ---------------------------------------------------------------------------
extern "C" void kernel_launch(void* const* d_in, const int* in_sizes, int n_in,
                              void* d_out, int out_size, void* d_ws, size_t ws_size,
                              hipStream_t stream) {
  (void)in_sizes; (void)n_in; (void)out_size; (void)ws_size;
  const float*     x    = (const float*)d_in[0];
  const long long* ei   = (const long long*)d_in[1];
  const float*     ew   = (const float*)d_in[2];
  // d_in[3] = batch (unused by reference)
  const float* c1w0 = (const float*)d_in[4];
  const float* c1w1 = (const float*)d_in[5];
  const float* c1b  = (const float*)d_in[6];
  const float* c2w0 = (const float*)d_in[7];
  const float* c2w1 = (const float*)d_in[8];
  const float* c2b  = (const float*)d_in[9];
  const float* l1wih = (const float*)d_in[10];
  const float* l1whh = (const float*)d_in[11];
  const float* l1b   = (const float*)d_in[12];
  const float* l2wih = (const float*)d_in[13];
  const float* l2whh = (const float*)d_in[14];
  const float* l2b   = (const float*)d_in[15];
  const float* l3wih = (const float*)d_in[16];
  const float* l3whh = (const float*)d_in[17];
  const float* l3b   = (const float*)d_in[18];
  const float* fc1w = (const float*)d_in[19];
  const float* fc1b = (const float*)d_in[20];
  const float* fc2w = (const float*)d_in[21];
  const float* fc2b = (const float*)d_in[22];
  const float* fc3w = (const float*)d_in[23];
  const float* fc3b = (const float*)d_in[24];
  const float* fc4w = (const float*)d_in[25];
  const float* fc4b = (const float*)d_in[26];
  float* out = (float*)d_out;

  // bump allocator over d_ws
  char* ws = (char*)d_ws;
  size_t off = 0;
  auto alloc = [&](size_t bytes) -> void* {
    off = (off + 255) & ~(size_t)255;
    void* p = ws + off;
    off += bytes;
    return p;
  };
  float*  xt    = (float*)alloc((size_t)TDIM * NNODES * FDIM * 4);
  float*  dinv  = (float*)alloc((size_t)TDIM * NNODES * 4);
  float*  tx1   = (float*)alloc((size_t)TDIM * NNODES * FDIM * 4);
  float*  h1    = (float*)alloc((size_t)TDIM * NNODES * FDIM * 4);
  float*  tx2   = (float*)alloc((size_t)TDIM * NNODES * FDIM * 4);
  __bf16* xcb   = (__bf16*)alloc((size_t)BDIM * SEQ * 64 * 2);   // stride-64 padded
  __bf16* hseq1 = (__bf16*)alloc((size_t)BDIM * SEQ * 64 * 2);
  __bf16* hseq2 = (__bf16*)alloc((size_t)BDIM * SEQ * 128 * 2);
  __bf16* hseq3 = (__bf16*)alloc((size_t)BDIM * SEQ * 256 * 2);
  __bf16* hbuf  = (__bf16*)alloc((size_t)2 * BDIM * 256 * 2);
  unsigned* sync = (unsigned*)alloc(256);
  __bf16* Wb1 = (__bf16*)alloc((size_t)128 * 256 * 2);
  __bf16* Wb2 = (__bf16*)alloc((size_t)192 * 512 * 2);
  __bf16* Wb3 = (__bf16*)alloc((size_t)384 * 1024 * 2);
  __bf16* Wc1 = (__bf16*)alloc(512 * 2);
  __bf16* Wc2 = (__bf16*)alloc(512 * 2);
  __bf16* Wf1 = (__bf16*)alloc((size_t)FC1K * 256 * 2);
  __bf16* Wf2 = (__bf16*)alloc((size_t)256 * 128 * 2);
  __bf16* Wf3 = (__bf16*)alloc((size_t)128 * 64 * 2);
  __bf16* Wf4 = (__bf16*)alloc((size_t)64 * 16 * 2);
  float*  part = (float*)alloc((size_t)32 * 256 * 256 * 4);
  __bf16* z1  = (__bf16*)alloc((size_t)256 * 256 * 2);
  __bf16* z2  = (__bf16*)alloc((size_t)256 * 128 * 2);
  __bf16* z3  = (__bf16*)alloc((size_t)256 * 64 * 2);

  const int TB = 256;

  // ---- weight packing (independent) ----
  k_pack_cheb<<<2, TB, 0, stream>>>(c1w0, c1w1, Wc1);
  k_pack_cheb<<<2, TB, 0, stream>>>(c2w0, c2w1, Wc2);
  k_pack_lstm_frag<<<(128 * 256 + TB - 1) / TB, TB, 0, stream>>>(l1wih, l1whh, Wb1, 62, 64, 64);
  k_pack_lstm_frag<<<(192 * 512 + TB - 1) / TB, TB, 0, stream>>>(l2wih, l2whh, Wb2, 64, 64, 128);
  k_pack_lstm_frag<<<(384 * 1024 + TB - 1) / TB, TB, 0, stream>>>(l3wih, l3whh, Wb3, 128, 128, 256);
  k_pack_frag<<<((size_t)FC1K * 256 + TB - 1) / TB, TB, 0, stream>>>(fc1w, Wf1, FC1K, 256, 256);
  k_pack_frag<<<(256 * 128 + TB - 1) / TB, TB, 0, stream>>>(fc2w, Wf2, 256, 128, 128);
  k_pack_frag<<<(128 * 64 + TB - 1) / TB, TB, 0, stream>>>(fc3w, Wf3, 128, 64, 64);
  k_pack_frag<<<(64 * 16 + TB - 1) / TB, TB, 0, stream>>>(fc4w, Wf4, 64, 4, 16);

  // ---- graph preprocessing ----
  int nxt = NNODES * FDIM * TDIM;
  k_transpose_x<<<(nxt + TB - 1) / TB, TB, 0, stream>>>(x, xt);
  (void)hipMemsetAsync(dinv, 0, (size_t)TDIM * NNODES * 4, stream);
  int nte = TDIM * EDIM;
  k_deg<<<(nte + TB - 1) / TB, TB, 0, stream>>>(ei, ew, dinv);
  k_dinv<<<(TDIM * NNODES + TB - 1) / TB, TB, 0, stream>>>(dinv);

  // ---- Cheb layer 1 ----
  (void)hipMemsetAsync(tx1, 0, (size_t)TDIM * NNODES * FDIM * 4, stream);
  k_scatter<<<(nte + TB - 1) / TB, TB, 0, stream>>>(ei, ew, dinv, xt, tx1);
  k_cheb<<<(TDIM * (NNODES / 16)) / 8, TB, 0, stream>>>(xt, tx1, Wc1, c1b, 1, h1, nullptr);

  // ---- Cheb layer 2 -> xcb (zero first: pad cols must not feed NaN to WMMA) ----
  (void)hipMemsetAsync(xcb, 0, (size_t)BDIM * SEQ * 64 * 2, stream);
  (void)hipMemsetAsync(tx2, 0, (size_t)TDIM * NNODES * FDIM * 4, stream);
  k_scatter<<<(nte + TB - 1) / TB, TB, 0, stream>>>(ei, ew, dinv, h1, tx2);
  k_cheb<<<(TDIM * (NNODES / 16)) / 8, TB, 0, stream>>>(h1, tx2, Wc2, c2b, 2, nullptr, xcb);

  // ---- LSTM stack (persistent fused kernels) ----
  (void)hipMemsetAsync(hbuf, 0, (size_t)2 * BDIM * 256 * 2, stream);
  (void)hipMemsetAsync(sync, 0, 8, stream);
  k_lstm<<<8, TB, 0, stream>>>(xcb, Wb1, l1b, hbuf, hseq1, sync, 64, 64, 8);

  (void)hipMemsetAsync(hbuf, 0, (size_t)2 * BDIM * 256 * 2, stream);
  (void)hipMemsetAsync(sync, 0, 8, stream);
  k_lstm<<<16, TB, 0, stream>>>(hseq1, Wb2, l2b, hbuf, hseq2, sync, 64, 128, 16);

  (void)hipMemsetAsync(hbuf, 0, (size_t)2 * BDIM * 256 * 2, stream);
  (void)hipMemsetAsync(sync, 0, 8, stream);
  k_lstm<<<32, TB, 0, stream>>>(hseq2, Wb3, l3b, hbuf, hseq3, sync, 128, 256, 32);

  // ---- MLP head ----
  dim3 g1(16, 32);
  k_fc1<<<g1, TB, 0, stream>>>(hseq3, Wf1, part, 32);
  k_fc1_reduce<<<(256 * 256) / TB, TB, 0, stream>>>(part, fc1b, z1, 32);
  // fc2: 16x8 tiles = 128 waves
  k_gemm<<<16, TB, 0, stream>>>(z1, 256, Wf2, fc2b, 1, z2, nullptr, 128, 8, 128, 256);
  // fc3: 16x4 tiles = 64 waves
  k_gemm<<<8, TB, 0, stream>>>(z2, 128, Wf3, fc3b, 1, z3, nullptr, 64, 4, 64, 128);
  // fc4: 16x1 tiles = 16 waves, f32 out [256,4]
  k_gemm<<<2, TB, 0, stream>>>(z3, 64, Wf4, fc4b, 0, nullptr, out, 4, 1, 4, 64);
}
